// GatedSlotAttention2_83313775608201
// MI455X (gfx1250) — matrix operand order repaired and checked by reading, not compiled, not run on hardware
//
#include <hip/hip_runtime.h>
#include <hip/hip_bf16.h>
#include <math.h>

// ---------------------------------------------------------------------------
// Gated Slot Attention 2 forward — MI455X (gfx1250, wave32, WMMA)
// B=1, T=1024, HID=2048, H=16, DK=DV=128, M=128, KW=4
//
// Strategy (MI455X roofline):
//  * 5 big GEMMs (~43 of ~50 GFLOP) -> f16 WMMA, f32 accumulate.  All operands
//    pre-packed once to f16 (weights transposed to [N,K]); GEMM inner loop is
//    pure global_load_b128 fragment loads + v_wmma_f32_16x16x32_f16.
//    Each wave owns a 32x64 C tile: 8 WMMAs per k-step, B fragments reused
//    across 2 row tiles, A fragments across 4 col tiles (1.5 b128/WMMA).
//    Tile validity is a compile-time template property -> no EXEC juggling
//    around WMMA (which requires EXEC all-1s).
//  * Recurrence (T=1024 strictly sequential) -> per-head 128x128 Sk/Sv state
//    resident in VGPRs (64+64 f32 per thread, 256-thread block), wave32
//    shuffle softmax; streams only 5*128 floats/step.
// ---------------------------------------------------------------------------

#define T_LEN 1024
#define HID   2048
#define NH    16
#define DK    128
#define DV    128
#define NM    128
#define SCALE 0.08838834764831845f   /* 1/sqrt(128) */

typedef __attribute__((ext_vector_type(16))) _Float16 v16h;
typedef __attribute__((ext_vector_type(8)))  _Float16 v8h;
typedef __attribute__((ext_vector_type(8)))  float    v8f;

__device__ __forceinline__ float sigmf(float x) { return 1.0f / (1.0f + __expf(-x)); }

// ---------------------------------------------------------------------------
// f32 -> f16 pack (row-major copy)
// ---------------------------------------------------------------------------
__global__ void pack_f16(const float* __restrict__ x, _Float16* __restrict__ y, int n)
{
  const int i = blockIdx.x * blockDim.x + threadIdx.x;
  if (i < n) y[i] = (_Float16)x[i];
}

// ---------------------------------------------------------------------------
// f32 [K,N] -> f16 [N,K] transpose-pack (32x32 LDS tile, coalesced both ways)
// grid = (ceil(N/32), ceil(K/32)), block = 256 (32x8)
// ---------------------------------------------------------------------------
__global__ __launch_bounds__(256) void transpose_pack_f16(
    const float* __restrict__ B, _Float16* __restrict__ BT, int K, int N)
{
  __shared__ float tile[32][33];
  const int kb = blockIdx.y * 32, nb = blockIdx.x * 32;
  const int tx = threadIdx.x & 31, ty = threadIdx.x >> 5;
  for (int r = ty; r < 32; r += 8) {
    const int k = kb + r, n = nb + tx;
    tile[r][tx] = (k < K && n < N) ? B[(size_t)k * N + n] : 0.0f;
  }
  __syncthreads();
  for (int r = ty; r < 32; r += 8) {
    const int n = nb + r, k = kb + tx;
    if (n < N && k < K) BT[(size_t)n * K + k] = (_Float16)tile[tx][r];
  }
}

// ---------------------------------------------------------------------------
// GEMM: C[Mr,N] = A16[Mr,K] @ BT16[N,K]^T (+ bias), f16 operands, f32 acc.
// block = 256 = 8 waves; wave -> 32x64 C tile (2 row-tiles x 4 col-tiles).
// grid.x = ceil(N/512), grid.y = Mr/32.  K % 32 == 0.
// FULL=true: every active wave's 64-col span is fully inside N (N % 64 == 0),
// so there are no per-tile guards and WMMA always runs with EXEC all-1s.
// Fragment layouts per CDNA5 ISA 7.12.2 (16-bit A 16x32 / B 32x16, wave32):
//   A lane group g (0/1): elem e<8 -> K = k0+8g+e ; e>=8 -> K = k0+16+8g+(e-8)
//   B lane group g:       elem e<8 -> K = k0+8g+e ; e>=8 -> K = k0+16+8g+(e-8)
// ---------------------------------------------------------------------------
template <bool FULL>
__global__ __launch_bounds__(256) void gemm_wmma_f16(
    const _Float16* __restrict__ A16, const _Float16* __restrict__ BT16,
    const float* __restrict__ bias, float* __restrict__ C,
    int Mr, int K, int N)
{
  const int wave   = threadIdx.x >> 5;
  const int lane   = threadIdx.x & 31;
  const int laneLo = lane & 15;
  const int hiSel  = lane >> 4;

  const int row0    = blockIdx.y * 32;
  const int colBase = (blockIdx.x * 8 + wave) * 64;
  if (colBase >= N) return;

  const _Float16* __restrict__ arow0 = A16 + (size_t)(row0 + laneLo) * K;
  const _Float16* __restrict__ arow1 = arow0 + (size_t)16 * K;

  bool valid[4];
  const _Float16* brow[4];
#pragma unroll
  for (int j = 0; j < 4; ++j) {
    valid[j] = FULL || ((colBase + j * 16) < N);
    const int c = valid[j] ? (colBase + j * 16 + laneLo) : 0;
    brow[j] = BT16 + (size_t)c * K;
  }

  v8f acc[2][4] = {{{}, {}, {}, {}}, {{}, {}, {}, {}}};

  for (int k0 = 0; k0 < K; k0 += 32) {
    __builtin_prefetch(arow0 + k0 + 256, 0, 1);   // global_prefetch_b8
    const v8h a0_lo = *(const v8h*)(arow0 + k0 + hiSel * 8);
    const v8h a0_hi = *(const v8h*)(arow0 + k0 + 16 + hiSel * 8);
    const v8h a1_lo = *(const v8h*)(arow1 + k0 + hiSel * 8);
    const v8h a1_hi = *(const v8h*)(arow1 + k0 + 16 + hiSel * 8);
    const v16h af0 = __builtin_shufflevector(a0_lo, a0_hi,
        0, 1, 2, 3, 4, 5, 6, 7, 8, 9, 10, 11, 12, 13, 14, 15);
    const v16h af1 = __builtin_shufflevector(a1_lo, a1_hi,
        0, 1, 2, 3, 4, 5, 6, 7, 8, 9, 10, 11, 12, 13, 14, 15);
#pragma unroll
    for (int j = 0; j < 4; ++j) {
      if (!FULL && !valid[j]) continue;
      const v8h b_lo = *(const v8h*)(brow[j] + k0 + hiSel * 8);
      const v8h b_hi = *(const v8h*)(brow[j] + k0 + 16 + hiSel * 8);
      const v16h bf = __builtin_shufflevector(b_lo, b_hi,
          0, 1, 2, 3, 4, 5, 6, 7, 8, 9, 10, 11, 12, 13, 14, 15);
      acc[0][j] = __builtin_amdgcn_wmma_f32_16x16x32_f16(
          false, af0, false, bf, (short)0, acc[0][j], false, false);
      acc[1][j] = __builtin_amdgcn_wmma_f32_16x16x32_f16(
          false, af1, false, bf, (short)0, acc[1][j], false, false);
    }
  }

  // C/D 16x16 f32 layout: VGPR r, lane -> row = r + hiSel*8, col = laneLo
#pragma unroll
  for (int j = 0; j < 4; ++j) {
    if (!FULL && !valid[j]) continue;
    const int col = colBase + j * 16 + laneLo;
    const float bv = bias ? bias[col] : 0.0f;
#pragma unroll
    for (int rt = 0; rt < 2; ++rt) {
#pragma unroll
      for (int r = 0; r < 8; ++r)
        C[(size_t)(row0 + rt * 16 + r + hiSel * 8) * N + col] = acc[rt][j][r] + bv;
    }
  }
}

// ---------------------------------------------------------------------------
// Causal depthwise conv1d (K=4) + SiLU.  x,y: [T, C], w: [C, 4]
// ---------------------------------------------------------------------------
__global__ void conv4_silu(const float* __restrict__ x, const float* __restrict__ w,
                           float* __restrict__ y, int Tn, int C)
{
  const int idx = blockIdx.x * blockDim.x + threadIdx.x;
  if (idx >= Tn * C) return;
  const int t = idx / C;
  const int c = idx % C;
  float acc = 0.0f;
#pragma unroll
  for (int i = 0; i < 4; ++i) {
    const int tt = t - 3 + i;
    const float xv = (tt >= 0) ? x[(size_t)tt * C + c] : 0.0f;
    acc += xv * w[c * 4 + i];
  }
  y[idx] = acc * sigmf(acc);   // silu
}

__global__ void sigmoid_inplace(float* __restrict__ x, int n)
{
  const int i = blockIdx.x * blockDim.x + threadIdx.x;
  if (i < n) x[i] = sigmf(x[i]);
}

// ---------------------------------------------------------------------------
// Per (t,h) row: bw[m] = beta * w[m] * rsqrt(sum_m w^2 + 1e-6)
// ---------------------------------------------------------------------------
__global__ void wprep_l2_beta(const float* __restrict__ wconv,
                              const float* __restrict__ beta,
                              float* __restrict__ bw)
{
  const int row = blockIdx.x;          // t*H + h
  const int m   = threadIdx.x;         // 0..127
  const float wv = wconv[(size_t)row * NM + m];
  __shared__ float red[128];
  red[m] = wv * wv;
  __syncthreads();
  for (int s = 64; s > 0; s >>= 1) {
    if (m < s) red[m] += red[m + s];
    __syncthreads();
  }
  const float inv = rsqrtf(red[0] + 1e-6f);
  bw[(size_t)row * NM + m] = beta[row] * wv * inv;
}

// ---------------------------------------------------------------------------
// Gated RMS norm: on = o * rsqrt(mean_d o^2 + 1e-5) * norm_w[d] * sigmoid(gate)
// ---------------------------------------------------------------------------
__global__ void rmsnorm_gate(const float* __restrict__ o,
                             const float* __restrict__ gate,
                             const float* __restrict__ norm_w,
                             float* __restrict__ on)
{
  const int row = blockIdx.x;
  const int d   = threadIdx.x;
  const float ov = o[(size_t)row * DV + d];
  __shared__ float red[128];
  red[d] = ov * ov;
  __syncthreads();
  for (int s = 64; s > 0; s >>= 1) {
    if (d < s) red[d] += red[d + s];
    __syncthreads();
  }
  const float inv = rsqrtf(red[0] * (1.0f / (float)DV) + 1e-5f);
  const float gv  = gate[(size_t)row * DV + d];
  on[(size_t)row * DV + d] = ov * inv * norm_w[d] * sigmf(gv);
}

// ---------------------------------------------------------------------------
// Sequential gated-slot scan, one block per head, 256 threads (8 wave32).
// State in VGPRs: thread (half=tid>>7, idx=tid&127):
//   Sk: owns col m = idx, rows d in [half*64, half*64+64)
//   Sv: owns col d = idx, rows m in [half*64, half*64+64)
// ---------------------------------------------------------------------------
__global__ __launch_bounds__(256) void gsa2_scan(
    const float* __restrict__ q, const float* __restrict__ k,
    const float* __restrict__ v, const float* __restrict__ bw,
    const float* __restrict__ gt, float* __restrict__ o)
{
  const int head = blockIdx.x;     // 0..15
  const int tid  = threadIdx.x;
  const int half = tid >> 7;       // 0,1
  const int idx  = tid & 127;
  const int base = half * 64;

  float sk[64], sv[64];
#pragma unroll
  for (int i = 0; i < 64; ++i) { sk[i] = 0.0f; sv[i] = 0.0f; }

  __shared__ float sq[128], skv[128], svv[128], sgt[128], sbw[128];
  __shared__ float spart[2][128], opart[2][128], sa[128];
  __shared__ float smax, ssum;

  for (int t = 0; t < T_LEN; ++t) {
    const size_t off = ((size_t)t * NH + head) * 128;
    if (tid < 128) {
      sq[tid]  = q[off + tid];
      skv[tid] = k[off + tid];
      svv[tid] = v[off + tid];
    } else {
      sgt[idx] = gt[off + idx];
      sbw[idx] = bw[off + idx];
    }
    __syncthreads();

    // Sk[d][m] = Sk*gt[m] + k[d]*bw[m];   s_part[m] = sum_d q[d]*Sk[d][m]
    {
      const float gtm = sgt[idx];
      const float bwm = sbw[idx];
      float part = 0.0f;
#pragma unroll
      for (int i = 0; i < 64; ++i) {
        sk[i] = sk[i] * gtm + skv[base + i] * bwm;
        part += sq[base + i] * sk[i];
      }
      spart[half][idx] = part;
    }
    // Sv[m][d] = Sv*gt[m] + bw[m]*v[d]
    {
      const float vd = svv[idx];
#pragma unroll
      for (int i = 0; i < 64; ++i) {
        const int m2 = base + i;
        sv[i] = sv[i] * sgt[m2] + sbw[m2] * vd;
      }
    }
    __syncthreads();

    if (tid < 128) sa[tid] = (spart[0][tid] + spart[1][tid]) * SCALE;
    __syncthreads();

    // softmax over 128 slots (wave32 reductions)
    if (tid < 32) {
      float mx = -3.4e38f;
      for (int j = tid; j < 128; j += 32) mx = fmaxf(mx, sa[j]);
      for (int s = 16; s > 0; s >>= 1) mx = fmaxf(mx, __shfl_xor(mx, s, 32));
      if (tid == 0) smax = mx;
    }
    __syncthreads();
    if (tid < 128) sa[tid] = __expf(sa[tid] - smax);
    __syncthreads();
    if (tid < 32) {
      float sum = 0.0f;
      for (int j = tid; j < 128; j += 32) sum += sa[j];
      for (int s = 16; s > 0; s >>= 1) sum += __shfl_xor(sum, s, 32);
      if (tid == 0) ssum = sum;
    }
    __syncthreads();

    // o[d] = (1/sum) * sum_m exp(s[m]) * Sv[m][d]
    {
      const float inv = 1.0f / ssum;
      float part = 0.0f;
#pragma unroll
      for (int i = 0; i < 64; ++i) part += sa[base + i] * sv[i];
      opart[half][idx] = part * inv;
    }
    __syncthreads();
    if (tid < 128) o[off + tid] = opart[0][tid] + opart[1][tid];
    __syncthreads();
  }
}

// ---------------------------------------------------------------------------
// Host-side orchestration
// ---------------------------------------------------------------------------
extern "C" void kernel_launch(void* const* d_in, const int* in_sizes, int n_in,
                              void* d_out, int out_size, void* d_ws, size_t ws_size,
                              hipStream_t stream)
{
  const float* h      = (const float*)d_in[0];   // [T, HID]
  const float* Wq     = (const float*)d_in[1];   // [HID, 2048]
  const float* Wk     = (const float*)d_in[2];
  const float* Wv     = (const float*)d_in[3];
  const float* Ww     = (const float*)d_in[4];
  const float* cq     = (const float*)d_in[5];   // [2048, 4]
  const float* ck     = (const float*)d_in[6];
  const float* cv     = (const float*)d_in[7];
  const float* Wf1    = (const float*)d_in[8];   // [HID, 128]
  const float* Wf2    = (const float*)d_in[9];   // [128, 2048]
  const float* Wb     = (const float*)d_in[10];  // [HID, 16]
  const float* Wg1    = (const float*)d_in[11];  // [HID, 128]
  const float* Wg2    = (const float*)d_in[12];  // [128, 2048]
  const float* bg2    = (const float*)d_in[13];  // [2048]
  const float* norm_w = (const float*)d_in[14];  // [128]
  const float* Wo     = (const float*)d_in[15];  // [2048, HID]
  float* out = (float*)d_out;                    // [T, HID]

  // ---- f32 scratch region ----
  float* ws = (float*)d_ws;
  const size_t TS = (size_t)T_LEN * HID;         // 2,097,152
  float* qp    = ws + 0 * TS;
  float* kp    = ws + 1 * TS;
  float* vp    = ws + 2 * TS;
  float* wp    = ws + 3 * TS;
  float* qc    = ws + 4 * TS;
  float* kc    = ws + 5 * TS;
  float* vc    = ws + 6 * TS;
  float* wc    = ws + 7 * TS;
  float* gtb   = ws + 8 * TS;                    // h@Wf1@Wf2 -> sigmoid
  float* gate  = ws + 9 * TS;
  float* obuf  = ws + 10 * TS;
  float* onorm = ws + 11 * TS;
  float* bwb   = ws + 12 * TS;
  float* fbuf  = ws + 13 * TS;                   // [T, 128]
  float* g1buf = fbuf + (size_t)T_LEN * 128;
  float* bpre  = g1buf + (size_t)T_LEN * 128;    // [T, 16]
  float* f32_end = bpre + (size_t)T_LEN * NH;

  // ---- f16 scratch region (packed operands) ----
  _Float16* w16 = (_Float16*)f32_end;
  _Float16* h16     = w16;                 w16 += TS;             // [T, HID]
  _Float16* WqT     = w16;                 w16 += TS;             // [2048, HID]
  _Float16* WkT     = w16;                 w16 += TS;
  _Float16* WvT     = w16;                 w16 += TS;
  _Float16* WwT     = w16;                 w16 += TS;
  _Float16* WoT     = w16;                 w16 += TS;             // [HID, 2048]
  _Float16* Wf1T    = w16;                 w16 += (size_t)128 * HID;
  _Float16* WbT     = w16;                 w16 += (size_t)16 * HID;
  _Float16* Wg1T    = w16;                 w16 += (size_t)128 * HID;
  _Float16* Wf2T    = w16;                 w16 += (size_t)HID * 128;
  _Float16* Wg2T    = w16;                 w16 += (size_t)HID * 128;
  _Float16* f16buf  = w16;                 w16 += (size_t)T_LEN * 128;
  _Float16* g116    = w16;                 w16 += (size_t)T_LEN * 128;
  _Float16* onorm16 = w16;                 /* += TS */

  const dim3 blk(256);
  const int nTS = (int)TS;
  const dim3 gEW((nTS + 255) / 256);

  // 0) pack operands to f16 (weights transposed to [N,K])
  pack_f16<<<gEW, blk, 0, stream>>>(h, h16, nTS);
  {
    const dim3 gT2048(HID / 32, HID / 32);                    // [2048 x 2048]
    transpose_pack_f16<<<gT2048, blk, 0, stream>>>(Wq, WqT, HID, HID);
    transpose_pack_f16<<<gT2048, blk, 0, stream>>>(Wk, WkT, HID, HID);
    transpose_pack_f16<<<gT2048, blk, 0, stream>>>(Wv, WvT, HID, HID);
    transpose_pack_f16<<<gT2048, blk, 0, stream>>>(Ww, WwT, HID, HID);
    transpose_pack_f16<<<gT2048, blk, 0, stream>>>(Wo, WoT, HID, HID);
    transpose_pack_f16<<<dim3(128 / 32, HID / 32), blk, 0, stream>>>(Wf1, Wf1T, HID, 128);
    transpose_pack_f16<<<dim3(1,        HID / 32), blk, 0, stream>>>(Wb,  WbT,  HID, 16);
    transpose_pack_f16<<<dim3(128 / 32, HID / 32), blk, 0, stream>>>(Wg1, Wg1T, HID, 128);
    transpose_pack_f16<<<dim3(HID / 32, 128 / 32), blk, 0, stream>>>(Wf2, Wf2T, 128, HID);
    transpose_pack_f16<<<dim3(HID / 32, 128 / 32), blk, 0, stream>>>(Wg2, Wg2T, 128, HID);
  }

  // grids: 32 rows per block-row now
  const dim3 gN2048((HID + 511) / 512, T_LEN / 32);   // (4, 32)
  const dim3 gN128(1, T_LEN / 32);                    // waves 0-1 active (full)
  const dim3 gN16(1, T_LEN / 32);                     // wave 0, guarded

  // 1) big projections (f16 WMMA, f32 accumulate) — FULL tiles, no guards
  gemm_wmma_f16<true><<<gN2048, blk, 0, stream>>>(h16, WqT, nullptr, qp, T_LEN, HID, HID);
  gemm_wmma_f16<true><<<gN2048, blk, 0, stream>>>(h16, WkT, nullptr, kp, T_LEN, HID, HID);
  gemm_wmma_f16<true><<<gN2048, blk, 0, stream>>>(h16, WvT, nullptr, vp, T_LEN, HID, HID);
  gemm_wmma_f16<true><<<gN2048, blk, 0, stream>>>(h16, WwT, nullptr, wp, T_LEN, HID, HID);

  // 2) small projections (N=128 full; N=16 guarded)
  gemm_wmma_f16<true ><<<gN128, blk, 0, stream>>>(h16, Wf1T, nullptr, fbuf,  T_LEN, HID, 128);
  gemm_wmma_f16<false><<<gN16,  blk, 0, stream>>>(h16, WbT,  nullptr, bpre,  T_LEN, HID, 16);
  gemm_wmma_f16<true ><<<gN128, blk, 0, stream>>>(h16, Wg1T, nullptr, g1buf, T_LEN, HID, 128);

  // 3) second-level projections (K=128)
  pack_f16<<<dim3((T_LEN * 128 + 255) / 256), blk, 0, stream>>>(fbuf,  f16buf, T_LEN * 128);
  pack_f16<<<dim3((T_LEN * 128 + 255) / 256), blk, 0, stream>>>(g1buf, g116,   T_LEN * 128);
  gemm_wmma_f16<true><<<gN2048, blk, 0, stream>>>(f16buf, Wf2T, nullptr, gtb,  T_LEN, 128, HID);
  gemm_wmma_f16<true><<<gN2048, blk, 0, stream>>>(g116,   Wg2T, bg2,     gate, T_LEN, 128, HID);

  // 4) causal conv + silu (reference applies cv to both v and w)
  conv4_silu<<<gEW, blk, 0, stream>>>(qp, cq, qc, T_LEN, HID);
  conv4_silu<<<gEW, blk, 0, stream>>>(kp, ck, kc, T_LEN, HID);
  conv4_silu<<<gEW, blk, 0, stream>>>(vp, cv, vc, T_LEN, HID);
  conv4_silu<<<gEW, blk, 0, stream>>>(wp, cv, wc, T_LEN, HID);

  // 5) gt = exp(logsigmoid(x)) = sigmoid(x); beta = sigmoid(bpre)
  sigmoid_inplace<<<gEW, blk, 0, stream>>>(gtb, nTS);
  sigmoid_inplace<<<dim3((T_LEN * NH + 255) / 256), blk, 0, stream>>>(bpre, T_LEN * NH);

  // 6) bw = beta * l2norm(w)
  wprep_l2_beta<<<dim3(T_LEN * NH), dim3(128), 0, stream>>>(wc, bpre, bwb);

  // 7) sequential gated-slot scan, one block per head
  gsa2_scan<<<dim3(NH), dim3(256), 0, stream>>>(qc, kc, vc, bwb, gtb, obuf);

  // 8) gated RMS norm
  rmsnorm_gate<<<dim3(T_LEN * NH), dim3(128), 0, stream>>>(obuf, gate, norm_w, onorm);

  // 9) output projection -> d_out
  pack_f16<<<gEW, blk, 0, stream>>>(onorm, onorm16, nTS);
  gemm_wmma_f16<true><<<gN2048, blk, 0, stream>>>(onorm16, WoT, nullptr, out, T_LEN, HID, HID);
}
